// Net_32169305047268
// MI455X (gfx1250) — compile-verified, matrix operand order checked by reading, not run on hardware
//
#include <hip/hip_runtime.h>
#include <hip/hip_bf16.h>

typedef __attribute__((ext_vector_type(16))) _Float16     v16h;
typedef __attribute__((ext_vector_type(8)))  float        v8f;
typedef __attribute__((ext_vector_type(4)))  unsigned int v4u;

#define GEMM_WAVES 4
#define GEMM_BLK   (GEMM_WAVES * 32)
#define LDS_PITCH  88   // halves; 176B row stride: 16B-aligned rows, conflict-free b128 reads

// ---------------------------------------------------------------------------
// WMMA GEMM: C[M,64] = A[M,K] @ B[K,64]   (f32 in, f16 WMMA, f32 accumulate)
// 4 waves / block, each wave owns a 16-row x 64-col tile.
// K staged in 64-wide chunks in LDS (f32->f16), zero-padded past K.
// B is stored TRANSPOSED in LDS so A and B fragments are both built from two
// contiguous 16-byte LDS reads (ds_load_b128) per the ISA 7.12.2 layout:
//   VGPR 0-3 <- halves kbase..kbase+7, VGPR 4-7 <- halves kbase+16..kbase+23
// ---------------------------------------------------------------------------
__global__ __launch_bounds__(GEMM_BLK) void k_gemm_wmma(
    const float* __restrict__ A, const float* __restrict__ B,
    float* __restrict__ C, int M, int K)
{
    __shared__ __align__(16) _Float16 sA[GEMM_WAVES][16][LDS_PITCH];
    __shared__ __align__(16) _Float16 sBT[64][LDS_PITCH];   // [n][k]

    const int tid  = threadIdx.x;
    const int lane = tid & 31;
    const int w    = tid >> 5;
    const int rowbase = blockIdx.x * (GEMM_WAVES * 16) + w * 16;

    v8f acc0 = {}, acc1 = {}, acc2 = {}, acc3 = {};

    const int nchunks = (K + 63) >> 6;
    for (int c = 0; c < nchunks; ++c) {
        const int k0 = c << 6;
        // stage B chunk [64k][64n] -> sBT[n][k]  (global reads coalesced over n)
        #pragma unroll 4
        for (int i = 0; i < 32; ++i) {
            int idx = tid + i * GEMM_BLK;      // 0..4095
            int kl = idx >> 6, n = idx & 63;
            float v = 0.f;
            if (k0 + kl < K) v = B[(size_t)(k0 + kl) * 64 + n];
            sBT[n][kl] = (_Float16)v;
        }
        // stage A chunk: 16 rows x 64 k per wave (coalesced over k)
        #pragma unroll 4
        for (int i = 0; i < 32; ++i) {
            int idx = lane + i * 32;           // 0..1023
            int r = idx >> 6, kl = idx & 63;
            int row = rowbase + r;
            float v = 0.f;
            if (row < M && (k0 + kl) < K) v = A[(size_t)row * K + k0 + kl];
            sA[w][r][kl] = (_Float16)v;
        }
        __syncthreads();

        const int half8 = (lane >> 4) << 3;    // 0 (lanes 0-15) or 8 (lanes 16-31)
        const int mrow  = lane & 15;
        const int ncol  = lane & 15;
        const _Float16* ap  = &sA[w][mrow][0];
        const _Float16* bp0 = &sBT[ 0 + ncol][0];
        const _Float16* bp1 = &sBT[16 + ncol][0];
        const _Float16* bp2 = &sBT[32 + ncol][0];
        const _Float16* bp3 = &sBT[48 + ncol][0];
        #pragma unroll
        for (int ks = 0; ks < 64; ks += 32) {
            const int kbase = ks + half8;      // byte offset 2*kbase: 16B-aligned
            union { v16h h; v4u q[2]; } af, bf0, bf1, bf2, bf3;
            af.q[0]  = *(const v4u*)(ap  + kbase);
            af.q[1]  = *(const v4u*)(ap  + kbase + 16);
            bf0.q[0] = *(const v4u*)(bp0 + kbase);
            bf0.q[1] = *(const v4u*)(bp0 + kbase + 16);
            bf1.q[0] = *(const v4u*)(bp1 + kbase);
            bf1.q[1] = *(const v4u*)(bp1 + kbase + 16);
            bf2.q[0] = *(const v4u*)(bp2 + kbase);
            bf2.q[1] = *(const v4u*)(bp2 + kbase + 16);
            bf3.q[0] = *(const v4u*)(bp3 + kbase);
            bf3.q[1] = *(const v4u*)(bp3 + kbase + 16);
            acc0 = __builtin_amdgcn_wmma_f32_16x16x32_f16(false, af.h, false, bf0.h, (short)0, acc0, false, false);
            acc1 = __builtin_amdgcn_wmma_f32_16x16x32_f16(false, af.h, false, bf1.h, (short)0, acc1, false, false);
            acc2 = __builtin_amdgcn_wmma_f32_16x16x32_f16(false, af.h, false, bf2.h, (short)0, acc2, false, false);
            acc3 = __builtin_amdgcn_wmma_f32_16x16x32_f16(false, af.h, false, bf3.h, (short)0, acc3, false, false);
        }
        __syncthreads();
    }

    // C/D layout: VGPR r -> row r (lanes 0-15) or row r+8 (lanes 16-31); N = lane&15
    #pragma unroll
    for (int r = 0; r < 8; ++r) {
        int row = rowbase + r + ((lane >> 4) << 3);
        if (row < M) {
            size_t base = (size_t)row * 64 + (lane & 15);
            C[base +  0] = acc0[r];
            C[base + 16] = acc1[r];
            C[base + 32] = acc2[r];
            C[base + 48] = acc3[r];
        }
    }
}

// ---------------------------------------------------------------------------
// small helper kernels
// ---------------------------------------------------------------------------
__global__ void k_fill(float* __restrict__ p, float v, int n) {
    int i = blockIdx.x * 256 + threadIdx.x;
    if (i < n) p[i] = v;
}

// e_src[n,h] = sum_d h[n, h*dim+d] * att_src[h,d]  (same for dst)
__global__ void k_scores(const float* __restrict__ h,
                         const float* __restrict__ asrc, const float* __restrict__ adst,
                         float* __restrict__ es, float* __restrict__ ed,
                         int N, int heads, int dim) {
    int t = blockIdx.x * 256 + threadIdx.x;
    if (t >= N * heads) return;
    int n = t / heads, hd = t - n * heads;
    const float* hp = h + (size_t)n * 64 + hd * dim;
    const float* as = asrc + hd * dim;
    const float* ad = adst + hd * dim;
    float s = 0.f, d = 0.f;
    for (int i = 0; i < dim; ++i) { float v = hp[i]; s += v * as[i]; d += v * ad[i]; }
    es[t] = s; ed[t] = d;
}

__device__ inline void atomicMaxF(float* addr, float val) {
    if (val >= 0.f) atomicMax((int*)addr, __float_as_int(val));
    else            atomicMin((unsigned int*)addr, __float_as_uint(val));
}

// pass 1: e = leaky_relu(es[src]+ed[dst]); store; segment max into mm[dst]
__global__ void k_edge_max(const int* __restrict__ srcs, const int* __restrict__ dsts,
                           int E, int N, int hshift,
                           const float* __restrict__ es, const float* __restrict__ ed,
                           float* __restrict__ ee, float* __restrict__ mm) {
    int t = blockIdx.x * 256 + threadIdx.x;
    const int heads = 1 << hshift;
    if (t >= (E + N) * heads) return;
    int e  = t >> hshift;
    int hd = t & (heads - 1);
    int s = (e < E) ? srcs[e] : (e - E);   // trailing self-loops
    int d = (e < E) ? dsts[e] : (e - E);
    float v = es[s * heads + hd] + ed[d * heads + hd];
    v = (v >= 0.f) ? v : 0.2f * v;
    ee[t] = v;
    atomicMaxF(&mm[d * heads + hd], v);
}

// pass 2: ex = exp(e - m[dst]); store; segment sum into dn[dst]
__global__ void k_edge_sum(const int* __restrict__ srcs, const int* __restrict__ dsts,
                           int E, int N, int hshift,
                           float* __restrict__ ee, const float* __restrict__ mm,
                           float* __restrict__ dn) {
    int t = blockIdx.x * 256 + threadIdx.x;
    const int heads = 1 << hshift;
    if (t >= (E + N) * heads) return;
    int e  = t >> hshift;
    int hd = t & (heads - 1);
    int d = (e < E) ? dsts[e] : (e - E);
    float ex = __expf(ee[t] - mm[d * heads + hd]);
    ee[t] = ex;
    atomicAdd(&dn[d * heads + hd], ex);
}

// pass 3: out[dst, d] += h[src, d] * (ex / denom[dst]) ; 64 lanes per edge, coalesced
__global__ void k_edge_aggr(const int* __restrict__ srcs, const int* __restrict__ dsts,
                            int E, int N, int hshift, int dshift,
                            const float* __restrict__ ee, const float* __restrict__ dn,
                            const float* __restrict__ h, float* __restrict__ out) {
    int t = blockIdx.x * 256 + threadIdx.x;
    if (t >= (E + N) * 64) return;
    const int heads = 1 << hshift;
    int e = t >> 6, d = t & 63;
    int s  = (e < E) ? srcs[e] : (e - E);
    int ds = (e < E) ? dsts[e] : (e - E);
    int hd = d >> dshift;
    float alpha = ee[e * heads + hd] / dn[ds * heads + hd];
    atomicAdd(&out[(size_t)ds * 64 + d], h[(size_t)s * 64 + d] * alpha);
}

__global__ void k_bias_elu(const float* __restrict__ acc, const float* __restrict__ b,
                           float* __restrict__ a, int N) {
    int t = blockIdx.x * 256 + threadIdx.x;
    if (t >= N * 64) return;
    float v = acc[t] + b[t & 63];
    a[t] = (v > 0.f) ? v : expm1f(v);
}

__global__ void k_bias_lsm(const float* __restrict__ acc, const float* __restrict__ b,
                           float* __restrict__ out, int N) {
    int n = blockIdx.x * 256 + threadIdx.x;
    if (n >= N) return;
    float row[64];
    float mx = -3.402823466e38f;
    for (int i = 0; i < 64; ++i) { row[i] = acc[(size_t)n * 64 + i] + b[i]; mx = fmaxf(mx, row[i]); }
    float sum = 0.f;
    for (int i = 0; i < 64; ++i) sum += __expf(row[i] - mx);
    float lse = mx + __logf(sum);
    for (int i = 0; i < 64; ++i) out[(size_t)n * 64 + i] = row[i] - lse;
}

// ---------------------------------------------------------------------------
extern "C" void kernel_launch(void* const* d_in, const int* in_sizes, int n_in,
                              void* d_out, int out_size, void* d_ws, size_t ws_size,
                              hipStream_t stream) {
    const float* x   = (const float*)d_in[0];
    const int*   ei  = (const int*)  d_in[1];
    const float* W1  = (const float*)d_in[2];
    const float* as1 = (const float*)d_in[3];
    const float* ad1 = (const float*)d_in[4];
    const float* b1  = (const float*)d_in[5];
    const float* W2  = (const float*)d_in[6];
    const float* as2 = (const float*)d_in[7];
    const float* ad2 = (const float*)d_in[8];
    const float* b2  = (const float*)d_in[9];

    const int Fin = 300;
    const int N = in_sizes[0] / Fin;
    const int E = in_sizes[1] / 2;
    const int* srcs = ei;
    const int* dsts = ei + E;

    float* ws = (float*)d_ws;
    size_t o = 0;
    float* h1   = ws + o; o += (size_t)N * 64;
    float* acc1 = ws + o; o += (size_t)N * 64;
    float* a1   = ws + o; o += (size_t)N * 64;
    float* h2   = ws + o; o += (size_t)N * 64;
    float* acc2 = ws + o; o += (size_t)N * 64;
    float* es   = ws + o; o += (size_t)N * 8;
    float* ed   = ws + o; o += (size_t)N * 8;
    float* mm   = ws + o; o += (size_t)N * 8;
    float* dn   = ws + o; o += (size_t)N * 8;
    float* ee   = ws + o; o += (size_t)(E + N) * 8;
    (void)ws_size; (void)n_in; (void)out_size;

    auto cdiv = [](int a, int b) { return (a + b - 1) / b; };
    const int EN = E + N;

    // -------- layer 1 (heads=8, dim=8) --------
    k_gemm_wmma<<<cdiv(N, 64), GEMM_BLK, 0, stream>>>(x, W1, h1, N, Fin);
    k_scores   <<<cdiv(N * 8, 256), 256, 0, stream>>>(h1, as1, ad1, es, ed, N, 8, 8);
    k_fill     <<<cdiv(N * 8, 256), 256, 0, stream>>>(mm, -3.402823466e38f, N * 8);
    k_fill     <<<cdiv(N * 8, 256), 256, 0, stream>>>(dn, 0.f, N * 8);
    k_fill     <<<cdiv(N * 64, 256), 256, 0, stream>>>(acc1, 0.f, N * 64);
    k_edge_max <<<cdiv(EN * 8, 256), 256, 0, stream>>>(srcs, dsts, E, N, 3, es, ed, ee, mm);
    k_edge_sum <<<cdiv(EN * 8, 256), 256, 0, stream>>>(srcs, dsts, E, N, 3, ee, mm, dn);
    k_edge_aggr<<<cdiv(EN * 64, 256), 256, 0, stream>>>(srcs, dsts, E, N, 3, 3, ee, dn, h1, acc1);
    k_bias_elu <<<cdiv(N * 64, 256), 256, 0, stream>>>(acc1, b1, a1, N);

    // -------- layer 2 (heads=1, dim=64) --------
    k_gemm_wmma<<<cdiv(N, 64), GEMM_BLK, 0, stream>>>(a1, W2, h2, N, 64);
    k_scores   <<<cdiv(N, 256), 256, 0, stream>>>(h2, as2, ad2, es, ed, N, 1, 64);
    k_fill     <<<cdiv(N, 256), 256, 0, stream>>>(mm, -3.402823466e38f, N);
    k_fill     <<<cdiv(N, 256), 256, 0, stream>>>(dn, 0.f, N);
    k_fill     <<<cdiv(N * 64, 256), 256, 0, stream>>>(acc2, 0.f, N * 64);
    k_edge_max <<<cdiv(EN, 256), 256, 0, stream>>>(srcs, dsts, E, N, 0, es, ed, ee, mm);
    k_edge_sum <<<cdiv(EN, 256), 256, 0, stream>>>(srcs, dsts, E, N, 0, ee, mm, dn);
    k_edge_aggr<<<cdiv(EN * 64, 256), 256, 0, stream>>>(srcs, dsts, E, N, 0, 6, ee, dn, h2, acc2);
    k_bias_lsm <<<cdiv(N, 256), 256, 0, stream>>>(acc2, b2, (float*)d_out, N);
}
